// NetGIN_38414187495858
// MI455X (gfx1250) — matrix operand order repaired and checked by reading, not compiled, not run on hardware
//
#include <hip/hip_runtime.h>
#include <math.h>

// ---------------- Problem constants (match reference) ----------------
#define NNODES 50000
#define NEDGES 1600000L
#define DDIM   128
#define FIN    1273
#define NGRAPH 512
#define BN_EPS 1e-5f

// ---------------- WMMA types (CDNA5 / gfx1250, wave32) ----------------
typedef __attribute__((ext_vector_type(16))) __bf16    v16bf;
typedef __attribute__((ext_vector_type(8)))  float     v8f;
typedef __attribute__((ext_vector_type(4)))  float     v4f;
typedef __attribute__((ext_vector_type(4)))  unsigned  v4u;
typedef __attribute__((ext_vector_type(8)))  unsigned  v8u;
typedef __attribute__((ext_vector_type(4)))  int       v4i;
typedef __attribute__((ext_vector_type(8)))  int       v8i;

#if defined(__has_builtin)
#if __has_builtin(__builtin_amdgcn_tensor_load_to_lds) && __has_builtin(__builtin_amdgcn_s_wait_tensorcnt)
#define HAVE_TDM 1
#endif
#endif
#ifndef HAVE_TDM
#define HAVE_TDM 0
#endif

static __device__ __forceinline__ unsigned short f2bfbits(float f) {
    // round-to-nearest-even fp32 -> bf16 bit pattern
    unsigned u = __builtin_bit_cast(unsigned, f);
    return (unsigned short)((u + 0x7FFFu + ((u >> 16) & 1u)) >> 16);
}
static __device__ __forceinline__ __bf16 f2bf(float f) {
    unsigned short s = f2bfbits(f);
    return __builtin_bit_cast(__bf16, s);
}

#if HAVE_TDM
typedef __attribute__((address_space(3))) void lds_void;
static __device__ __forceinline__ unsigned lds_addr_of(const void* p) {
    return (unsigned)(unsigned long long)(lds_void*)p;   // LDS byte offset
}
// Issue a 2D TDM tile load: tile (tile0 x tile1) of 4-byte elements from
// row-major tensor (line stride = stride0 elems), zero-filling reads past
// (td0, td1).  D# packing per CDNA5 ISA 8.3/8.4 (count=1, type=2).
static __device__ __forceinline__ void tdm_load_2d(unsigned ldsAddr,
        const float* gaddr, unsigned tile0, unsigned tile1,
        unsigned td0, unsigned td1, unsigned long long stride0)
{
    unsigned long long ga = (unsigned long long)gaddr;
    v4u g0;
    g0[0] = 1u;                                       // count=1 (valid)
    g0[1] = ldsAddr;                                  // lds_addr
    g0[2] = (unsigned)ga;                             // global_addr[31:0]
    g0[3] = (unsigned)((ga >> 32) & 0x1FFFFFFu) | (2u << 30);  // addr hi | type=2
    v8i g1;
    g1[0] = 0x20000;                                  // data_size=4B, wg_mask=0
    g1[1] = (int)((td0 & 0xFFFFu) << 16);             // tensor_dim0 lo
    g1[2] = (int)((td0 >> 16) | ((td1 & 0xFFFFu) << 16));
    g1[3] = (int)((td1 >> 16) | ((tile0 & 0xFFFFu) << 16));
    g1[4] = (int)(tile1 & 0xFFFFu);                   // tile_dim1 (tile_dim2=0)
    g1[5] = (int)(stride0 & 0xFFFFFFFFu);             // tensor_dim0_stride lo
    g1[6] = (int)((stride0 >> 32) & 0xFFFFu);         // stride hi (dim1_stride=0)
    g1[7] = 0;
    v4i z4 = {};
#if __has_include(<hip/amd_detail/amd_gfx1250_TDM.h>)
    v8i z8 = {};
    __builtin_amdgcn_tensor_load_to_lds(g0, g1, z4, z4, z8, 0);
#else
    __builtin_amdgcn_tensor_load_to_lds(g0, g1, z4, z4, 0);
#endif
}
#endif

// =====================================================================
// bf16-WMMA GEMM: C[M,Nc] = op(A[M,K] x B) (+bias) (+acc) (+relu)
// A fp32 row-major (lda); B fp32 row-major; transB: B(k,n) = B[n*ldb+k].
//
// Wide path (Nc % 128 == 0): one 256-thread block = 8 wave32 on a 16x128
// C stripe. Per 32-K step the B slice (32x128 fp32) is DMA'd into LDS by
// the Tensor Data Mover (double-buffered on TENSORcnt; K-tail zero-fill
// by descriptor dims), then converted once to bf16 Bs[n][k]; the shared
// 16x32 A tile is staged/converted cooperatively. Fragments are two
// contiguous ds_load_b128 per operand feeding v_wmma_f32_16x16x32_bf16.
// Narrow path (tiny head GEMM): per-wave clamped scalar loads + K tail.
// =====================================================================
__global__ void gemm_wmma(const float* __restrict__ A, int lda,
                          const float* __restrict__ B, int ldb,
                          float* __restrict__ C, int ldc,
                          int M, int Ncols, int K,
                          const float* __restrict__ bias,
                          int transB, int acc, int dorelu)
{
    __shared__ __align__(16) unsigned short As[16 * 32];    // As[m][k] bf16
    __shared__ __align__(16) unsigned short Bs[128 * 32];   // Bs[n][k] bf16
#if HAVE_TDM
    __shared__ __align__(16) float Braw[2][32 * 128];       // raw fp32 TDM dest
#endif

    int lane = threadIdx.x & 31;
    int hi = lane >> 4;      // lane group (0: lanes 0-15, 1: lanes 16-31)
    int lo = lane & 15;
    int tilesM = (M + 15) >> 4;

    if ((Ncols & 127) == 0) {
        // ================= wide path (LDS-staged) =================
        int widx   = threadIdx.x >> 5;
        long wave0 = (long)blockIdx.x * 8;
        int tilesN = Ncols >> 4;
        int tmB = (int)(wave0 / tilesN);
        if (tmB >= tilesM) return;                 // uniform across block
        int colbase = (int)(wave0 % tilesN) << 4;  // block covers 128 columns
        int m0   = tmB << 4;
        int nLoc = (widx << 4) + lo;               // 0..127 within stripe
        int nGlb = colbase + nLoc;

        v8f c = {};
        if (acc) {
#pragma unroll
            for (int r = 0; r < 8; ++r)
                c[r] = C[(long)min(m0 + r + hi * 8, M - 1) * ldc + nGlb];
        }

        const int nIter = (K + 31) >> 5;
#if HAVE_TDM
        if (widx == 0) {       // pre-issue slice 0 (wave-level DMA, EXEC ignored)
            unsigned kv = (unsigned)min(32, K);
            if (transB)
                tdm_load_2d(lds_addr_of(&Braw[0][0]), B + (long)colbase * ldb,
                            32, 128, kv, 128, (unsigned long long)ldb);
            else
                tdm_load_2d(lds_addr_of(&Braw[0][0]), B + colbase,
                            128, 32, 128, kv, (unsigned long long)ldb);
        }
#endif
        for (int it = 0; it < nIter; ++it) {
            int k0 = it << 5;
            __syncthreads();   // Bs/As reuse guard (prev wmma reads done)
#if HAVE_TDM
            if (widx == 0) {
                if (it + 1 < nIter) {   // software pipeline: issue next, wait cur
                    int k1 = k0 + 32;
                    unsigned kv = (unsigned)min(32, K - k1);
                    float* nb = &Braw[(it + 1) & 1][0];
                    if (transB)
                        tdm_load_2d(lds_addr_of(nb), B + (long)colbase * ldb + k1,
                                    32, 128, kv, 128, (unsigned long long)ldb);
                    else
                        tdm_load_2d(lds_addr_of(nb), B + (long)k1 * ldb + colbase,
                                    128, 32, 128, kv, (unsigned long long)ldb);
                    __builtin_amdgcn_s_wait_tensorcnt(1);
                } else {
                    __builtin_amdgcn_s_wait_tensorcnt(0);
                }
            }
#endif
            // ---- stage A tile: 16x32, 2 elements/thread, coalesced in k ----
            for (int j = threadIdx.x; j < 512; j += 256) {
                int m = j >> 5, k = j & 31;
                int gk = k0 + k;
                const float* ap = A + (long)min(m0 + m, M - 1) * lda;
                __builtin_prefetch(ap + gk + 32, 0, 1);     // next k-slice
                float v = ap[min(gk, K - 1)];
                As[(m << 5) + k] = f2bfbits(gk < K ? v : 0.0f);
            }
#if HAVE_TDM
            {   // ---- convert TDM'd raw slice -> Bs[n][k] bf16 ----
                const float* braw = &Braw[it & 1][0];
                if (transB) {
                    for (int j = threadIdx.x; j < 4096; j += 256)
                        Bs[j] = f2bfbits(braw[j]);          // already [n][k]
                } else {
                    for (int j = threadIdx.x; j < 4096; j += 256) {
                        int n = j & 127, k = j >> 7;
                        Bs[(n << 5) + k] = f2bfbits(braw[(k << 7) + n]);
                    }
                }
            }
#else
            // ---- stage B slice from global: 32x128 -> Bs[n][k] ----
            for (int j = threadIdx.x; j < 4096; j += 256) {
                int n, k;
                if (transB) { k = j & 31;  n = j >> 5; }
                else        { n = j & 127; k = j >> 7; }
                int gk = k0 + k;
                int gn = colbase + n;
                float v = transB ? B[(long)gn * ldb + min(gk, K - 1)]
                                 : B[(long)min(gk, K - 1) * ldb + gn];
                Bs[(n << 5) + k] = f2bfbits(gk < K ? v : 0.0f);
            }
#endif
            __syncthreads();
            // ---- fragments: contiguous 16B LDS chunks -> ds_load_b128 ----
            const unsigned short* ap = As + (lo << 5) + (hi << 3);
            v4u a0 = *(const v4u*)ap;
            v4u a1 = *(const v4u*)(ap + 16);
            const unsigned short* bp = Bs + (nLoc << 5) + (hi << 3);
            v4u b0 = *(const v4u*)bp;
            v4u b1 = *(const v4u*)(bp + 16);
            v8u ar, br;
#pragma unroll
            for (int q = 0; q < 4; ++q) {
                ar[q] = a0[q]; ar[q + 4] = a1[q];
                br[q] = b0[q]; br[q + 4] = b1[q];
            }
            v16bf a = __builtin_bit_cast(v16bf, ar);
            v16bf b = __builtin_bit_cast(v16bf, br);
            c = __builtin_amdgcn_wmma_f32_16x16x32_bf16(
                    false, a, false, b, (short)0, c, false, false);
        }

        float bv = bias ? bias[nGlb] : 0.0f;
        if (m0 + 16 <= M) {
#pragma unroll
            for (int r = 0; r < 8; ++r) {
                float v = c[r] + bv;
                if (dorelu) v = fmaxf(v, 0.0f);
                C[(long)(m0 + r + hi * 8) * ldc + nGlb] = v;
            }
        } else {
#pragma unroll
            for (int r = 0; r < 8; ++r) {
                int m = m0 + r + hi * 8;
                if (m < M) {
                    float v = c[r] + bv;
                    if (dorelu) v = fmaxf(v, 0.0f);
                    C[(long)m * ldc + nGlb] = v;
                }
            }
        }
        return;
    }

    // ================= narrow fallback (per-wave, tiny GEMMs) =================
    int wave  = blockIdx.x * (blockDim.x >> 5) + (threadIdx.x >> 5);
    int tilesN = (Ncols + 15) >> 4;
    if (wave >= tilesM * tilesN) return;           // wave-uniform
    int tm = wave / tilesN, tn = wave - tm * tilesN;
    int m0 = tm << 4, n0 = tn << 4;

    int am = min(m0 + lo, M - 1);                  // address clamp policy
    int bn = min(n0 + lo, Ncols - 1);
    const float* Arow = A + (long)am * lda;

    v8f c = {};
    if (acc) {
#pragma unroll
        for (int r = 0; r < 8; ++r)
            c[r] = C[(long)min(m0 + r + hi * 8, M - 1) * ldc + bn];
    }

    const int kmain = K & ~31;
    const int kA0 = hi << 4;
    const int kB0 = hi << 4;

    for (int k0 = 0; k0 < kmain; k0 += 32) {
        __builtin_prefetch(Arow + k0 + 256, 0, 1);
        v16bf a, b;
#pragma unroll
        for (int e = 0; e < 8; ++e) {
            a[e]     = f2bf(Arow[k0 + kA0 + e]);
            a[e + 8] = f2bf(Arow[k0 + kA0 + 8 + e]);
        }
        if (transB) {
            const float* Brow = B + (long)bn * ldb + k0 + kB0;
#pragma unroll
            for (int e = 0; e < 16; ++e) b[e] = f2bf(Brow[e]);
        } else {
            const float* Bcol = B + (long)(k0 + kB0) * ldb + bn;
#pragma unroll
            for (int e = 0; e < 16; ++e) b[e] = f2bf(Bcol[(long)e * ldb]);
        }
        c = __builtin_amdgcn_wmma_f32_16x16x32_bf16(
                false, a, false, b, (short)0, c, false, false);
    }
    if (kmain < K) {     // K tail: clamped address + value select
        v16bf a, b;
#pragma unroll
        for (int e = 0; e < 16; ++e) {
            int k = kmain + kA0 + e;
            float v = Arow[min(k, K - 1)];
            a[e] = f2bf(k < K ? v : 0.0f);
        }
#pragma unroll
        for (int e = 0; e < 16; ++e) {
            int k = kmain + kB0 + e;
            int kc = min(k, K - 1);
            float v = transB ? B[(long)bn * ldb + kc] : B[(long)kc * ldb + bn];
            b[e] = f2bf(k < K ? v : 0.0f);
        }
        c = __builtin_amdgcn_wmma_f32_16x16x32_bf16(
                false, a, false, b, (short)0, c, false, false);
    }

    float bv = bias ? bias[bn] : 0.0f;
    int nIdx = n0 + lo;
    if (nIdx < Ncols) {
#pragma unroll
        for (int r = 0; r < 8; ++r) {
            int m = m0 + r + hi * 8;
            if (m < M) {
                float v = c[r] + bv;
                if (dorelu) v = fmaxf(v, 0.0f);
                C[(long)m * ldc + nIdx] = v;
            }
        }
    }
}

// ---------------- s = (1+eps)*y + b1   (pre-aggregation init) ----------------
__global__ void gin_init(const float* __restrict__ y, float* __restrict__ s,
                         const float* __restrict__ epsp,
                         const float* __restrict__ b1, long n)
{
    long i = (long)blockIdx.x * blockDim.x + threadIdx.x;
    if (i >= n) return;
    int d = (int)(i & (DDIM - 1));
    s[i] = (1.0f + epsp[0]) * y[i] + b1[d];
}

// ---------------- edge scatter-add: s[dst] += y[src], 4 floats/thread ----------------
__global__ void edge_agg(const int* __restrict__ ei, long E,
                         const float* __restrict__ y, float* __restrict__ s)
{
    long t = (long)blockIdx.x * blockDim.x + threadIdx.x;
    if (t >= E * (DDIM / 4)) return;
    long e = t >> 5;                       // DDIM/4 == 32 groups per edge
    int  d4 = (int)(t & 31) << 2;
    int src = ei[e];
    int dst = ei[E + e];
    v4f v = *(const v4f*)(y + (long)src * DDIM + d4);
    float* sd = s + (long)dst * DDIM + d4;
    atomicAdd(sd + 0, v[0]);
    atomicAdd(sd + 1, v[1]);
    atomicAdd(sd + 2, v[2]);
    atomicAdd(sd + 3, v[3]);
}

// ---------------- column sum / sumsq for BatchNorm (128 cols) ----------------
__global__ void colsum(const float* __restrict__ s, int Nrows,
                       float* __restrict__ sums)   // sums[0..127], sumsq[128..255]
{
    int d = threadIdx.x;                 // blockDim.x == 128
    int r0 = blockIdx.x * 256;
    int rend = min(r0 + 256, Nrows);
    float a = 0.0f, b = 0.0f;
    for (int r = r0; r < rend; ++r) {
        float v = s[(long)r * DDIM + d];
        a += v; b += v * v;
    }
    atomicAdd(&sums[d], a);
    atomicAdd(&sums[DDIM + d], b);
}

// ---------------- BN (batch stats) + ReLU, strided column output ----------------
__global__ void bn_relu(const float* __restrict__ s, const float* __restrict__ sums,
                        int Nrows, const float* __restrict__ g,
                        const float* __restrict__ bb,
                        float* __restrict__ out, int out_ld, int col0)
{
    long i = (long)blockIdx.x * blockDim.x + threadIdx.x;
    if (i >= (long)Nrows * DDIM) return;
    int d = (int)(i & (DDIM - 1));
    long r = i >> 7;
    float invN = 1.0f / (float)Nrows;
    float mu  = sums[d] * invN;
    float var = sums[DDIM + d] * invN - mu * mu;
    float v = g[d] * (s[i] - mu) * rsqrtf(var + BN_EPS) + bb[d];
    out[r * out_ld + col0 + d] = fmaxf(v, 0.0f);
}

// ---------------- misc small kernels ----------------
__global__ void fillk(float* __restrict__ p, float v, long n) {
    long i = (long)blockIdx.x * blockDim.x + threadIdx.x;
    if (i < n) p[i] = v;
}
__global__ void addvec(const float* a, const float* b, float* o, int n) {
    int i = blockIdx.x * blockDim.x + threadIdx.x;
    if (i < n) o[i] = a[i] + b[i];
}

// ---------------- LSTM pointwise (gate order i,f,g,o) ----------------
__global__ void lstm_pt(const float* __restrict__ gates,
                        float* __restrict__ hh, float* __restrict__ cc, int n)
{
    int i = blockIdx.x * blockDim.x + threadIdx.x;
    if (i >= n) return;
    int g = i >> 7, d = i & (DDIM - 1);
    const float* row = gates + (long)g * (4 * DDIM);
    float ig = row[d], fg = row[DDIM + d], gg = row[2 * DDIM + d], og = row[3 * DDIM + d];
    float si = 1.0f / (1.0f + expf(-ig));
    float sf = 1.0f / (1.0f + expf(-fg));
    float so = 1.0f / (1.0f + expf(-og));
    float cv = sf * cc[i] + si * tanhf(gg);
    cc[i] = cv;
    hh[i] = so * tanhf(cv);
}

// ---------------- e[n] = dot(h[n,:], hh[batch[n],:]) — one wave32 per node ----------------
__global__ void dote(const float* __restrict__ h, const float* __restrict__ hh,
                     const int* __restrict__ batch, int Nn, float* __restrict__ e)
{
    int w = blockIdx.x * (blockDim.x >> 5) + (threadIdx.x >> 5);
    int lane = threadIdx.x & 31;
    if (w >= Nn) return;
    int g = batch[w];
    const float* hr = h  + (long)w * DDIM;
    const float* qr = hh + (long)g * DDIM;
    float acc = 0.0f;
    for (int d = lane; d < DDIM; d += 32) acc += hr[d] * qr[d];
#pragma unroll
    for (int off = 16; off > 0; off >>= 1) acc += __shfl_xor(acc, off, 32);
    if (lane == 0) e[w] = acc;
}

static __device__ __forceinline__ int lowerb(const int* a, int n, int key) {
    int lo = 0, hi = n;
    while (lo < hi) { int mid = (lo + hi) >> 1; if (a[mid] < key) lo = mid + 1; else hi = mid; }
    return lo;
}

// ---------------- per-graph softmax + weighted sum r[g,:] (batch is sorted) ----------------
__global__ void seg_attn(const float* __restrict__ h, const int* __restrict__ batch,
                         float* __restrict__ e, int Nn, float* __restrict__ r)
{
    int g = blockIdx.x;
    int lo  = lowerb(batch, Nn, g);
    int hi2 = lowerb(batch, Nn, g + 1);
    __shared__ float red[256];
    __shared__ float s_emax, s_den;
    int t = threadIdx.x;
    // 1) max
    float mx = -3.4e38f;
    for (int i = lo + t; i < hi2; i += 256) mx = fmaxf(mx, e[i]);
    red[t] = mx; __syncthreads();
    for (int s = 128; s > 0; s >>= 1) { if (t < s) red[t] = fmaxf(red[t], red[t + s]); __syncthreads(); }
    if (t == 0) s_emax = red[0];
    __syncthreads();
    // 2) exp + denom (in-place into e; node ranges are disjoint per block)
    float sum = 0.0f;
    for (int i = lo + t; i < hi2; i += 256) { float ex = expf(e[i] - s_emax); e[i] = ex; sum += ex; }
    red[t] = sum; __syncthreads();
    for (int s = 128; s > 0; s >>= 1) { if (t < s) red[t] += red[t + s]; __syncthreads(); }
    if (t == 0) s_den = red[0];
    __syncthreads();
    float inv = (hi2 > lo && s_den != 0.0f) ? 1.0f / s_den : 0.0f;
    // 3) r[g,d] = sum_n a[n]*h[n,d]
    for (int d = t; d < DDIM; d += 256) {
        float acc = 0.0f;
        for (int i = lo; i < hi2; ++i) acc += e[i] * h[(long)i * DDIM + d];
        r[(long)g * DDIM + d] = acc * inv;
    }
}

__global__ void qstar_upd(const float* __restrict__ hh, const float* __restrict__ r,
                          float* __restrict__ qs, int n)
{
    int i = blockIdx.x * blockDim.x + threadIdx.x;
    if (i >= n) return;
    int g = i >> 7, d = i & (DDIM - 1);
    qs[(long)g * (2 * DDIM) + d]        = hh[i];
    qs[(long)g * (2 * DDIM) + DDIM + d] = r[i];
}

// =====================================================================
// Host-side orchestration
// =====================================================================
static inline long cdivl(long a, long b) { return (a + b - 1) / b; }

static inline void run_gemm(const float* A, int lda, const float* B, int ldb,
                            float* C, int ldc, int M, int Nc, int K,
                            const float* bias, int transB, int acc, int dorelu,
                            hipStream_t stream)
{
    long tiles = cdivl(M, 16) * cdivl(Nc, 16);
    long blocks = cdivl(tiles, 8);             // 8 wave32 per 256-thread block
    gemm_wmma<<<dim3((unsigned)blocks), dim3(256), 0, stream>>>(
        A, lda, B, ldb, C, ldc, M, Nc, K, bias, transB, acc, dorelu);
}

extern "C" void kernel_launch(void* const* d_in, const int* in_sizes, int n_in,
                              void* d_out, int out_size, void* d_ws, size_t ws_size,
                              hipStream_t stream)
{
    const int   N = NNODES, G = NGRAPH;
    const long  E = NEDGES;

    const float* x        = (const float*)d_in[0];
    const int*   ei[3]    = { (const int*)d_in[1], (const int*)d_in[2], (const int*)d_in[3] };
    const int*   batch    = (const int*)d_in[4];
    const float* c1_W1    = (const float*)d_in[5];
    const float* c1_b1    = (const float*)d_in[6];
    const float* cr_W1    = (const float*)d_in[7];
    const float* cr_b1    = (const float*)d_in[8];
    const float* c_W2     = (const float*)d_in[9];
    const float* c_b2     = (const float*)d_in[10];
    const float* bn1_g    = (const float*)d_in[11];
    const float* bn1_b    = (const float*)d_in[12];
    const float* bn2_g    = (const float*)d_in[13];
    const float* bn2_b    = (const float*)d_in[14];
    const float* c_eps    = (const float*)d_in[15];
    const float* m_W1     = (const float*)d_in[16];
    const float* m_b1     = (const float*)d_in[17];
    const float* mbn1_g   = (const float*)d_in[18];
    const float* mbn1_b   = (const float*)d_in[19];
    const float* m_W2     = (const float*)d_in[20];
    const float* m_b2     = (const float*)d_in[21];
    const float* mbn2_g   = (const float*)d_in[22];
    const float* mbn2_b   = (const float*)d_in[23];
    const float* Wih      = (const float*)d_in[24];
    const float* Whh      = (const float*)d_in[25];
    const float* bih      = (const float*)d_in[26];
    const float* bhh      = (const float*)d_in[27];
    const float* fc1_W    = (const float*)d_in[28];
    const float* fc1_b    = (const float*)d_in[29];
    const float* fc4_W    = (const float*)d_in[30];
    const float* fc4_b    = (const float*)d_in[31];

    // ---- workspace carve-up (floats) ----
    float* ws = (float*)d_ws;
    size_t off = 0;
    auto carve = [&](size_t nf) { float* p = ws + off; off += nf; return p; };
    float* w_y   = carve((size_t)N * DDIM);       // GEMM out / BN-normalized temp
    float* w_s   = carve((size_t)N * DDIM);       // pre-BN buffer
    float* w_h   = carve((size_t)N * DDIM);       // layer output
    float* w_xc  = carve((size_t)N * 3 * DDIM);   // concat of 3 edge-set branches
    float* w_e   = carve((size_t)N);              // attention logits/weights
    float* w_st  = carve(256);                    // BN stats: sums[128], sumsq[128]
    float* w_qs  = carve((size_t)G * 2 * DDIM);
    float* w_hh  = carve((size_t)G * DDIM);
    float* w_cc  = carve((size_t)G * DDIM);
    float* w_gt  = carve((size_t)G * 4 * DDIM);
    float* w_r   = carve((size_t)G * DDIM);
    float* w_bs  = carve(4 * DDIM);
    float* w_t   = carve((size_t)G * DDIM);

    const long nND      = (long)N * DDIM;
    const long gridND   = cdivl(nND, 256);
    const long gridEdge = cdivl(E * (DDIM / 4), 256);
    const long gridCol  = cdivl(N, 256);

    // ================= 6 GIN layers =================
    for (int l = 0; l < 6; ++l) {
        for (int k = 0; k < 3; ++k) {
            const float* A   = (l == 0) ? x : w_h;
            const int    Kd  = (l == 0) ? FIN : DDIM;
            const float* W1  = (l == 0) ? (c1_W1 + (size_t)k * FIN * DDIM)
                                        : (cr_W1 + (size_t)((l - 1) * 3 + k) * DDIM * DDIM);
            const float* b1  = (l == 0) ? (c1_b1 + (size_t)k * DDIM)
                                        : (cr_b1 + (size_t)((l - 1) * 3 + k) * DDIM);
            const float* W2  = c_W2 + (size_t)(l * 3 + k) * DDIM * DDIM;
            const float* b2  = c_b2 + (size_t)(l * 3 + k) * DDIM;
            const float* g1  = bn1_g + (size_t)(l * 3 + k) * DDIM;
            const float* gb1 = bn1_b + (size_t)(l * 3 + k) * DDIM;
            const float* g2  = bn2_g + (size_t)(l * 3 + k) * DDIM;
            const float* gb2 = bn2_b + (size_t)(l * 3 + k) * DDIM;
            const float* ep  = c_eps + (size_t)(l * 3 + k);

            // y = h @ W1  (linearity: agg(h)@W1 == agg(h@W1) -> aggregate at width 128)
            run_gemm(A, Kd, W1, DDIM, w_y, DDIM, N, DDIM, Kd, nullptr, 0, 0, 0, stream);
            // s = (1+eps)*y + b1 ; s[dst] += y[src] over edge set k
            gin_init<<<dim3((unsigned)gridND), 256, 0, stream>>>(w_y, w_s, ep, b1, nND);
            edge_agg<<<dim3((unsigned)gridEdge), 256, 0, stream>>>(ei[k], E, w_y, w_s);
            // BN1 + ReLU
            fillk<<<1, 256, 0, stream>>>(w_st, 0.0f, 256);
            colsum<<<dim3((unsigned)gridCol), 128, 0, stream>>>(w_s, N, w_st);
            bn_relu<<<dim3((unsigned)gridND), 256, 0, stream>>>(w_s, w_st, N, g1, gb1, w_y, DDIM, 0);
            // second linear + BN2 + ReLU -> column block k of xcat
            run_gemm(w_y, DDIM, W2, DDIM, w_s, DDIM, N, DDIM, DDIM, b2, 0, 0, 0, stream);
            fillk<<<1, 256, 0, stream>>>(w_st, 0.0f, 256);
            colsum<<<dim3((unsigned)gridCol), 128, 0, stream>>>(w_s, N, w_st);
            bn_relu<<<dim3((unsigned)gridND), 256, 0, stream>>>(w_s, w_st, N, g2, gb2,
                                                                w_xc, 3 * DDIM, k * DDIM);
        }
        // fusion MLP: [N,384] -> 128 -> 128
        run_gemm(w_xc, 3 * DDIM, m_W1 + (size_t)l * 3 * DDIM * DDIM, DDIM,
                 w_y, DDIM, N, DDIM, 3 * DDIM, m_b1 + (size_t)l * DDIM, 0, 0, 0, stream);
        fillk<<<1, 256, 0, stream>>>(w_st, 0.0f, 256);
        colsum<<<dim3((unsigned)gridCol), 128, 0, stream>>>(w_y, N, w_st);
        bn_relu<<<dim3((unsigned)gridND), 256, 0, stream>>>(w_y, w_st, N,
                 mbn1_g + (size_t)l * DDIM, mbn1_b + (size_t)l * DDIM, w_s, DDIM, 0);
        run_gemm(w_s, DDIM, m_W2 + (size_t)l * DDIM * DDIM, DDIM,
                 w_y, DDIM, N, DDIM, DDIM, m_b2 + (size_t)l * DDIM, 0, 0, 0, stream);
        fillk<<<1, 256, 0, stream>>>(w_st, 0.0f, 256);
        colsum<<<dim3((unsigned)gridCol), 128, 0, stream>>>(w_y, N, w_st);
        bn_relu<<<dim3((unsigned)gridND), 256, 0, stream>>>(w_y, w_st, N,
                 mbn2_g + (size_t)l * DDIM, mbn2_b + (size_t)l * DDIM, w_h, DDIM, 0);
    }

    // ================= Set2Set readout =================
    fillk<<<dim3((unsigned)cdivl(G * 2 * DDIM, 256)), 256, 0, stream>>>(w_qs, 0.0f, (long)G * 2 * DDIM);
    fillk<<<dim3((unsigned)cdivl(G * DDIM, 256)), 256, 0, stream>>>(w_hh, 0.0f, (long)G * DDIM);
    fillk<<<dim3((unsigned)cdivl(G * DDIM, 256)), 256, 0, stream>>>(w_cc, 0.0f, (long)G * DDIM);
    addvec<<<2, 256, 0, stream>>>(bih, bhh, w_bs, 4 * DDIM);

    const long gridGD = cdivl((long)G * DDIM, 256);
    for (int step = 0; step < 6; ++step) {
        // gates = q_star @ Wih^T + hh @ Whh^T + (bih+bhh)
        run_gemm(w_qs, 2 * DDIM, Wih, 2 * DDIM, w_gt, 4 * DDIM,
                 G, 4 * DDIM, 2 * DDIM, w_bs, /*transB=*/1, /*acc=*/0, 0, stream);
        run_gemm(w_hh, DDIM, Whh, DDIM, w_gt, 4 * DDIM,
                 G, 4 * DDIM, DDIM, nullptr, /*transB=*/1, /*acc=*/1, 0, stream);
        lstm_pt<<<dim3((unsigned)gridGD), 256, 0, stream>>>(w_gt, w_hh, w_cc, G * DDIM);
        // attention logits, per-graph softmax, weighted readout
        dote<<<dim3((unsigned)cdivl(N, 8)), 256, 0, stream>>>(w_h, w_hh, batch, N, w_e);
        seg_attn<<<dim3(G), 256, 0, stream>>>(w_h, batch, w_e, N, w_r);
        qstar_upd<<<dim3((unsigned)gridGD), 256, 0, stream>>>(w_hh, w_r, w_qs, G * DDIM);
    }

    // ================= output head =================
    run_gemm(w_qs, 2 * DDIM, fc1_W, DDIM, w_t, DDIM, G, DDIM, 2 * DDIM,
             fc1_b, 0, 0, /*relu=*/1, stream);
    run_gemm(w_t, DDIM, fc4_W, 12, (float*)d_out, 12, G, 12, DDIM,
             fc4_b, 0, 0, 0, stream);
}
